// ModularFusedMoEKernel_17059610099907
// MI455X (gfx1250) — matrix operand order repaired and checked by reading, not compiled, not run on hardware
//
#include <hip/hip_runtime.h>
#include <stdint.h>

// ---------------- problem constants (from reference) ----------------
#define E_      8
#define T_      2048
#define D_      2048
#define DFF_    1024
#define K_TOP   2

// ---------------- GEMM tiling ----------------
#define BM      64          // block M tile (rows of routed slots)
#define BN      256         // block N tile (8 waves x 32 cols)
#define BK      32          // K step (one bf16 WMMA K)
#define CAP     4608        // padded routed-slot capacity: 4096 + 8*64, multiple of BM
#define MTILES  (CAP / BM)  // 72
#define LDS_ROW 5           // uint4 per LDS row: 64B data + 16B pad (conflict-free b128)

typedef __attribute__((ext_vector_type(16))) __bf16 v16bf;
typedef __attribute__((ext_vector_type(8)))  float  v8f;

union Frag16 { uint4 q[2]; v16bf v; };

#if __has_builtin(__builtin_amdgcn_global_load_async_to_lds_b128)
#define HAVE_ASYNC_LDS 1
// builtin signature (from hipcc diagnostic): (int4 AS(1)*, int4 AS(3)*, imm offset, imm cpol)
typedef int v4i_gcc __attribute__((__vector_size__(16)));
typedef __attribute__((address_space(1))) v4i_gcc* gas1_v4i;
typedef __attribute__((address_space(3))) v4i_gcc* las3_v4i;
__device__ __forceinline__ void async_copy16(const void* g, void* l) {
    __builtin_amdgcn_global_load_async_to_lds_b128(
        (gas1_v4i)(uintptr_t)g, (las3_v4i)(uint32_t)(uintptr_t)l, 0, 0);
}
__device__ __forceinline__ void wait_async0() {
#if __has_builtin(__builtin_amdgcn_s_wait_asynccnt)
    __builtin_amdgcn_s_wait_asynccnt(0);
#else
    asm volatile("s_wait_asynccnt 0x0" ::: "memory");
#endif
}
#endif

__device__ __forceinline__ unsigned short f32_to_bf16(float f) {
    union { float f; uint32_t u; } c; c.f = f;
    uint32_t u = c.u;
    uint32_t r = u + 0x7FFFu + ((u >> 16) & 1u);   // round-to-nearest-even
    return (unsigned short)(r >> 16);
}

// ---------------- fp32 -> bf16 conversion (vectorized, grid-stride) ----------------
__global__ void cvt_f32_to_bf16(const float* __restrict__ src,
                                unsigned short* __restrict__ dst, long n4) {
    long i      = (long)blockIdx.x * blockDim.x + threadIdx.x;
    long stride = (long)gridDim.x * blockDim.x;
    for (long j = i; j < n4; j += stride) {
        float4 f = *((const float4*)src + j);
        ushort4 o;
        o.x = f32_to_bf16(f.x); o.y = f32_to_bf16(f.y);
        o.z = f32_to_bf16(f.z); o.w = f32_to_bf16(f.w);
        *((ushort4*)dst + j) = o;
    }
}

// ---------------- deterministic routing (counting sort, one block) ----------------
__global__ void route_kernel(const int* __restrict__ ids,     // [T_*K_TOP]
                             int* __restrict__ routed,        // [CAP] slot -> token (-1 = pad)
                             int* __restrict__ slot_of,       // [T_*K_TOP] (t,k) -> slot
                             int* __restrict__ basep) {       // [E_+1] padded bases
    __shared__ int cnt[E_];
    __shared__ int base[E_ + 1];
    int t = threadIdx.x;
    if (t < E_) {
        int c = 0;
        for (int i = 0; i < T_ * K_TOP; ++i) c += (ids[i] == t);
        cnt[t] = c;
    }
    __syncthreads();
    if (t == 0) {
        int acc = 0;
        for (int e = 0; e < E_; ++e) { base[e] = acc; acc += ((cnt[e] + BM - 1) / BM) * BM; }
        base[E_] = acc;
    }
    __syncthreads();
    if (t < E_) {
        int s = base[t];
        for (int i = 0; i < T_ * K_TOP; ++i) {
            if (ids[i] == t) { routed[s] = i / K_TOP; slot_of[i] = s; ++s; }
        }
        for (; s < base[t + 1]; ++s) routed[s] = -1;  // pad rows (outputs never read)
    }
    __syncthreads();
    if (t == 0) for (int e = 0; e <= E_; ++e) basep[e] = base[e];
}

// ---------------- grouped bf16 GEMM: Out[m, n] = sum_k A[m,k] * W_e[n,k] ----------------
// A: bf16, ld = K. GATHER: A row = routed[m] (token), else A row = m (slot).
// W: bf16, per-expert [N x K], expert stride N*K. Out: fp32 [CAP x N].
template <bool GATHER>
__global__ void grouped_gemm_bf16(const unsigned short* __restrict__ A,
                                  const unsigned short* __restrict__ W,
                                  float* __restrict__ Out,
                                  const int* __restrict__ routed,
                                  const int* __restrict__ basep,
                                  int N, int K) {
    const int m0    = blockIdx.x * BM;
    const int total = basep[E_];
    if (m0 >= total) return;

    int e = 0;
    while (e < E_ && basep[e + 1] <= m0) ++e;
    const unsigned short* We = W + (size_t)e * N * K;

    // double-buffered A tile: 2 x 64 rows x 80B
    __shared__ uint4 ldsA[2 * BM * LDS_ROW];

    const int tid    = threadIdx.x;
    const int wave   = tid >> 5;
    const int lane   = tid & 31;
    const int laneLo = lane & 15;   // column / row-in-16
    const int laneHi = lane >> 4;   // K-half selector
    const int nWave  = blockIdx.y * BN + wave * 32;

    // staging: 256 threads x 16B = 64 rows x 64B per K-step
    const int srow   = tid >> 2;    // 0..63
    const int schunk = tid & 3;     // 0..3 (16B chunks)
    int arow = m0 + srow;
    if (GATHER) {
        int tok = routed[m0 + srow];
        arow = (tok >= 0) ? tok : 0;   // pad rows load token 0; results unused
    }
    const unsigned short* Arow = A + (size_t)arow * K;

    v8f acc[4][2];
    const v8f vzero = {0.f, 0.f, 0.f, 0.f, 0.f, 0.f, 0.f, 0.f};
#pragma unroll
    for (int i = 0; i < 4; ++i)
#pragma unroll
        for (int j = 0; j < 2; ++j) acc[i][j] = vzero;

#if defined(HAVE_ASYNC_LDS)
    // ---- async-to-LDS double-buffered pipeline: 1 barrier per K-step ----
    async_copy16((const uint4*)(Arow) + schunk,
                 &ldsA[srow * LDS_ROW + schunk]);
    int cur = 0;
    for (int k0 = 0; k0 < K; k0 += BK) {
        wait_async0();                 // my async write to LDS landed
        __syncthreads();               // everyone's landed; prev-buffer reads all done
        if (k0 + BK < K) {
            async_copy16((const uint4*)(Arow + k0 + BK) + schunk,
                         &ldsA[(cur ^ 1) * (BM * LDS_ROW) + srow * LDS_ROW + schunk]);
        }
        const uint4* buf = &ldsA[cur * (BM * LDS_ROW)];

        Frag16 b[2];
#pragma unroll
        for (int ns = 0; ns < 2; ++ns) {
            const unsigned short* wp =
                We + (size_t)(nWave + ns * 16 + laneLo) * K + k0 + laneHi * 16;
            b[ns].q[0] = *(const uint4*)(wp);
            b[ns].q[1] = *(const uint4*)(wp + 8);
        }
#pragma unroll
        for (int ms = 0; ms < 4; ++ms) {
            Frag16 a;
            const int r = ms * 16 + laneLo;
            a.q[0] = buf[r * LDS_ROW + 2 * laneHi];
            a.q[1] = buf[r * LDS_ROW + 2 * laneHi + 1];
#pragma unroll
            for (int ns = 0; ns < 2; ++ns) {
                acc[ms][ns] = __builtin_amdgcn_wmma_f32_16x16x32_bf16(
                    false, a.v, false, b[ns].v, (short)0, acc[ms][ns], false, false);
            }
        }
        cur ^= 1;
    }
#else
    // ---- fallback: synchronous staging through VGPRs ----
    for (int k0 = 0; k0 < K; k0 += BK) {
        uint4 av = *((const uint4*)(Arow + k0) + schunk);
        __syncthreads();
        ldsA[srow * LDS_ROW + schunk] = av;
        __syncthreads();

        Frag16 b[2];
#pragma unroll
        for (int ns = 0; ns < 2; ++ns) {
            const unsigned short* wp =
                We + (size_t)(nWave + ns * 16 + laneLo) * K + k0 + laneHi * 16;
            b[ns].q[0] = *(const uint4*)(wp);
            b[ns].q[1] = *(const uint4*)(wp + 8);
        }
#pragma unroll
        for (int ms = 0; ms < 4; ++ms) {
            Frag16 a;
            const int r = ms * 16 + laneLo;
            a.q[0] = ldsA[r * LDS_ROW + 2 * laneHi];
            a.q[1] = ldsA[r * LDS_ROW + 2 * laneHi + 1];
#pragma unroll
            for (int ns = 0; ns < 2; ++ns) {
                acc[ms][ns] = __builtin_amdgcn_wmma_f32_16x16x32_bf16(
                    false, a.v, false, b[ns].v, (short)0, acc[ms][ns], false, false);
            }
        }
    }
#endif

    // ---- epilogue: C/D layout -> row = (lane>=16)*8 + vgpr, col = lane%16 ----
#pragma unroll
    for (int ms = 0; ms < 4; ++ms) {
        const int row = m0 + ms * 16 + laneHi * 8;
#pragma unroll
        for (int ns = 0; ns < 2; ++ns) {
            const int col = nWave + ns * 16 + laneLo;
            float* op = Out + (size_t)row * N + col;
#pragma unroll
            for (int v = 0; v < 8; ++v) op[(size_t)v * N] = acc[ms][ns][v];
        }
    }
}

// ---------------- SiLU(gate) * up -> bf16 h ----------------
__global__ void silu_mul_kernel(const float* __restrict__ gu,
                                unsigned short* __restrict__ h,
                                const int* __restrict__ basep) {
    const int total = basep[E_];
    const long n = (long)total * DFF_;
    long i      = (long)blockIdx.x * blockDim.x + threadIdx.x;
    long stride = (long)gridDim.x * blockDim.x;
    for (; i < n; i += stride) {
        long s = i / DFF_;
        long c = i - s * DFF_;
        float g = gu[s * (2 * DFF_) + c];
        float u = gu[s * (2 * DFF_) + DFF_ + c];
        float sv = g / (1.f + __expf(-g));
        h[s * DFF_ + c] = f32_to_bf16(sv * u);
    }
}

// ---------------- combine: out[t] = w0*y[slot0] + w1*y[slot1] ----------------
__global__ void combine_kernel(const float* __restrict__ y,
                               const float* __restrict__ tw,
                               const int* __restrict__ slot_of,
                               float* __restrict__ out) {
    long i = (long)blockIdx.x * blockDim.x + threadIdx.x;   // float4 index
    long n4 = (long)T_ * D_ / 4;
    if (i >= n4) return;
    long idx = i * 4;
    int t = (int)(idx / D_);
    int d = (int)(idx - (long)t * D_);
    int s0 = slot_of[t * K_TOP + 0];
    int s1 = slot_of[t * K_TOP + 1];
    float w0 = tw[t * K_TOP + 0];
    float w1 = tw[t * K_TOP + 1];
    float4 a = *(const float4*)(y + (size_t)s0 * D_ + d);
    float4 b = *(const float4*)(y + (size_t)s1 * D_ + d);
    float4 o;
    o.x = w0 * a.x + w1 * b.x;
    o.y = w0 * a.y + w1 * b.y;
    o.z = w0 * a.z + w1 * b.z;
    o.w = w0 * a.w + w1 * b.w;
    *(float4*)(out + idx) = o;
}

// ---------------- workspace layout ----------------
static constexpr size_t alignup(size_t x) { return (x + 255) & ~(size_t)255; }
static constexpr size_t SZ_XB   = (size_t)T_ * D_ * 2;               // 8 MiB bf16 x
static constexpr size_t SZ_W1B  = (size_t)E_ * 2 * DFF_ * D_ * 2;    // 64 MiB bf16 w1
static constexpr size_t SZ_W2B  = (size_t)E_ * D_ * DFF_ * 2;        // 32 MiB bf16 w2
static constexpr size_t SZ_RT   = (size_t)CAP * 4;
static constexpr size_t SZ_SLOT = (size_t)T_ * K_TOP * 4;
static constexpr size_t SZ_BASE = 256;
static constexpr size_t SZ_GU   = (size_t)CAP * 2 * DFF_ * 4;        // fp32 gu, reused as y

static constexpr size_t OFF_XB   = 0;
static constexpr size_t OFF_W1B  = alignup(OFF_XB + SZ_XB);
static constexpr size_t OFF_W2B  = alignup(OFF_W1B + SZ_W1B);
static constexpr size_t OFF_RT   = alignup(OFF_W2B + SZ_W2B);
static constexpr size_t OFF_SLOT = alignup(OFF_RT + SZ_RT);
static constexpr size_t OFF_BASE = alignup(OFF_SLOT + SZ_SLOT);
static constexpr size_t OFF_GU   = alignup(OFF_BASE + SZ_BASE);
static constexpr size_t OFF_H    = alignup(OFF_GU + SZ_GU);

extern "C" void kernel_launch(void* const* d_in, const int* in_sizes, int n_in,
                              void* d_out, int out_size, void* d_ws, size_t ws_size,
                              hipStream_t stream) {
    (void)in_sizes; (void)n_in; (void)out_size; (void)ws_size;

    const float* x   = (const float*)d_in[0];   // [T, D]
    const float* w1  = (const float*)d_in[1];   // [E, 2*DFF, D]
    const float* w2  = (const float*)d_in[2];   // [E, D, DFF]
    const float* tw  = (const float*)d_in[3];   // [T, K]
    const int*   ids = (const int*)d_in[4];     // [T, K]
    float*       out = (float*)d_out;           // [T, D]

    char* ws = (char*)d_ws;
    unsigned short* xb   = (unsigned short*)(ws + OFF_XB);
    unsigned short* w1b  = (unsigned short*)(ws + OFF_W1B);
    unsigned short* w2b  = (unsigned short*)(ws + OFF_W2B);
    int*            rt   = (int*)(ws + OFF_RT);
    int*            slot = (int*)(ws + OFF_SLOT);
    int*            base = (int*)(ws + OFF_BASE);
    float*          gu   = (float*)(ws + OFF_GU);   // also reused as y
    unsigned short* h    = (unsigned short*)(ws + OFF_H);
    float*          y    = gu;                      // GEMM2 overwrites gu (no longer needed)

    // 1) fp32 -> bf16 (bf16 weights fit in 192MB L2; WMMA bf16 rate ~8x f32 WMMA)
    {
        long n4x  = (long)T_ * D_ / 4;
        long n4w1 = (long)E_ * 2 * DFF_ * D_ / 4;
        long n4w2 = (long)E_ * D_ * DFF_ / 4;
        cvt_f32_to_bf16<<<8192, 256, 0, stream>>>(x,  xb,  n4x);
        cvt_f32_to_bf16<<<8192, 256, 0, stream>>>(w1, w1b, n4w1);
        cvt_f32_to_bf16<<<8192, 256, 0, stream>>>(w2, w2b, n4w2);
    }

    // 2) deterministic routing
    route_kernel<<<1, 32, 0, stream>>>(ids, rt, slot, base);

    // 3) GEMM1: gu[slot, 0:2DFF] = x[token] @ w1[e]^T   (gathered A)
    grouped_gemm_bf16<true><<<dim3(MTILES, (2 * DFF_) / BN), 256, 0, stream>>>(
        xb, w1b, gu, rt, base, 2 * DFF_, D_);

    // 4) h = silu(gate) * up  (bf16)
    silu_mul_kernel<<<(CAP * DFF_) / 256, 256, 0, stream>>>(gu, h, base);

    // 5) GEMM2: y[slot, 0:D] = h[slot] @ w2[e]^T
    grouped_gemm_bf16<false><<<dim3(MTILES, D_ / BN), 256, 0, stream>>>(
        h, w2b, y, rt, base, D_, DFF_);

    // 6) combine with routing weights (exactly 2 contributions -> deterministic)
    combine_kernel<<<((long)T_ * D_ / 4 + 255) / 256, 256, 0, stream>>>(y, tw, slot, out);
}
